// BasicTransformerBlock_76527727280543
// MI455X (gfx1250) — compile-verified, matrix operand order checked by reading, not run on hardware
//
#include <hip/hip_runtime.h>
#include <math.h>

// ---------------- problem constants ----------------
constexpr int kBF    = 8;
constexpr int kSEQ   = 1024;
constexpr int kDIM   = 640;
constexpr int kHEADS = 8;
constexpr int kHD    = 80;
constexpr int kCTX   = 77;
constexpr int kCROSS = 768;
constexpr int kFFH   = 2560;          // half of GEGLU projection (4*DIM)
constexpr int kROWS  = kBF * kSEQ;    // 8192
constexpr int kLDVC  = 96;            // padded key stride for transposed cross-V

typedef __attribute__((ext_vector_type(16))) __bf16 v16bf;
typedef __attribute__((ext_vector_type(8)))  float  v8f;

struct U32x8 { uint4 a, b; };   // 32 bytes, same size as v16bf

// ---------------- scalar helpers ----------------
__device__ inline __bf16 us2bf(unsigned short s) { return __builtin_bit_cast(__bf16, s); }

__device__ inline unsigned short f2bf(float f) {           // round-to-nearest-even
  unsigned int u = __builtin_bit_cast(unsigned int, f);
  unsigned int r = (u + 0x7fffu + ((u >> 16) & 1u)) >> 16;
  return (unsigned short)r;
}

__device__ inline v8f v8f_zero() {
  v8f z;
#pragma unroll
  for (int i = 0; i < 8; ++i) z[i] = 0.f;
  return z;
}

__device__ inline float rmax16(float x) {                  // reduce over 16-lane half
  x = fmaxf(x, __shfl_xor(x, 1, 32));
  x = fmaxf(x, __shfl_xor(x, 2, 32));
  x = fmaxf(x, __shfl_xor(x, 4, 32));
  x = fmaxf(x, __shfl_xor(x, 8, 32));
  return x;
}
__device__ inline float rsum16(float x) {
  x += __shfl_xor(x, 1, 32);
  x += __shfl_xor(x, 2, 32);
  x += __shfl_xor(x, 4, 32);
  x += __shfl_xor(x, 8, 32);
  return x;
}

// ---------------- async copy to LDS (CDNA5, ASYNCcnt-tracked) ----------------
__device__ inline void async_copy_b128(unsigned lds_off, const unsigned short* g) {
  asm volatile("global_load_async_to_lds_b128 %0, %1, off"
               :: "v"(lds_off), "v"(g) : "memory");
}
__device__ inline unsigned lds_off_of(const void* p) {
  return (unsigned)(size_t)p;
}
__device__ inline void wait_async0() {
  asm volatile("s_wait_asynccnt 0x0" ::: "memory");
}

// ---------------- WMMA fragment helpers (ISA 7.12.2 layouts) ----------------
__device__ inline int frag_k0(int v, int hf) {
  return (v < 4 ? 2 * v : 16 + 2 * (v - 4)) + hf * 8;
}

// FAST A/B fragment (16x32, 16-bit) from row-major matrix, no bounds checks.
__device__ inline v16bf load_frag_fast(const unsigned short* p, int lda) {
  const int lane = threadIdx.x & 31;
  const int hf   = lane >> 4;
  const int m    = lane & 15;
  const unsigned short* row = p + (size_t)m * lda + hf * 8;
  U32x8 s;
  s.a = *(const uint4*)(row);        // k = hf*8 .. hf*8+7
  s.b = *(const uint4*)(row + 16);   // k = 16+hf*8 .. 16+hf*8+7
  return __builtin_bit_cast(v16bf, s);
}

// FAST fragment when only the first 16 K-columns are valid (head-dim tail).
__device__ inline v16bf load_frag_c2(const unsigned short* p, int lda) {
  const int lane = threadIdx.x & 31;
  const int hf   = lane >> 4;
  const int m    = lane & 15;
  U32x8 s;
  s.a = *(const uint4*)(p + (size_t)m * lda + hf * 8);
  s.b.x = 0u; s.b.y = 0u; s.b.z = 0u; s.b.w = 0u;
  return __builtin_bit_cast(v16bf, s);
}

// GUARDED fragment, branch-free; caller guarantees row 0 of p is in-bounds.
__device__ inline v16bf load_frag_guard(const unsigned short* p, int lda, int rows, int cols) {
  const int lane = threadIdx.x & 31;
  const int hf   = lane >> 4;
  const int m    = lane & 15;
  const int mc   = (m < rows) ? m : 0;
  const unsigned short* row = p + (size_t)mc * lda;
  v16bf f;
#pragma unroll
  for (int v = 0; v < 8; ++v) {
    int k0  = frag_k0(v, hf);
    int k0c = (k0 < cols) ? k0 : 0;
    int k1c = (k0 + 1 < cols) ? k0 + 1 : 0;
    unsigned short lo = row[k0c];
    unsigned short hi = row[k1c];
    f[2 * v]     = us2bf((m < rows && k0 < cols)     ? lo : (unsigned short)0);
    f[2 * v + 1] = us2bf((m < rows && k0 + 1 < cols) ? hi : (unsigned short)0);
  }
  return f;
}

__device__ inline v8f wmma_bf16(v16bf a, v16bf b, v8f c) {
  return __builtin_amdgcn_wmma_f32_16x16x32_bf16(false, a, false, b, (short)0, c, false, false);
}

// ---------------- utility kernels ----------------
__global__ __launch_bounds__(256) void wt_kernel(const float* __restrict__ W,
                                                 unsigned short* __restrict__ Wt,
                                                 int K, int N) {
  int idx = blockIdx.x * 256 + threadIdx.x;
  if (idx >= K * N) return;
  int n = idx / K, k = idx - n * K;
  Wt[idx] = f2bf(W[(size_t)k * N + n]);
}

__global__ __launch_bounds__(256) void cvt_kernel(const float* __restrict__ X,
                                                  unsigned short* __restrict__ O, int n) {
  int i = blockIdx.x * 256 + threadIdx.x;
  if (i < n) O[i] = f2bf(X[i]);
}

// transpose self-attn V: vbf (kROWS x kDIM) -> Vt [kBF*kHEADS][kHD][kSEQ]
__global__ __launch_bounds__(256) void vt_self_kernel(const unsigned short* __restrict__ V,
                                                      unsigned short* __restrict__ Vt) {
  size_t idx = (size_t)blockIdx.x * 256 + threadIdx.x;
  if (idx >= (size_t)kBF * kHEADS * kHD * kSEQ) return;
  int    s  = (int)(idx % kSEQ);
  size_t r  = idx / kSEQ;
  int    d  = (int)(r % kHD);
  size_t fh = r / kHD;
  int head  = (int)(fh % kHEADS);
  int bf    = (int)(fh / kHEADS);
  Vt[idx] = V[((size_t)bf * kSEQ + s) * kDIM + head * kHD + d];
}

// transpose cross-attn V: vbf (kBF*kCTX x kDIM) -> Vt [kBF*kHEADS][kHD][kLDVC], zero-padded
__global__ __launch_bounds__(256) void vt_cross_kernel(const unsigned short* __restrict__ V,
                                                       unsigned short* __restrict__ Vt) {
  size_t idx = (size_t)blockIdx.x * 256 + threadIdx.x;
  if (idx >= (size_t)kBF * kHEADS * kHD * kLDVC) return;
  int    s  = (int)(idx % kLDVC);
  size_t r  = idx / kLDVC;
  int    d  = (int)(r % kHD);
  size_t fh = r / kHD;
  int head  = (int)(fh % kHEADS);
  int bf    = (int)(fh / kHEADS);
  Vt[idx] = (s < kCTX) ? V[((size_t)bf * kCTX + s) * kDIM + head * kHD + d]
                       : (unsigned short)0;
}

// LayerNorm over rows of kDIM, f32 in -> bf16 out
__global__ __launch_bounds__(256) void ln_kernel(const float* __restrict__ X,
                                                 const float* __restrict__ g,
                                                 const float* __restrict__ b,
                                                 unsigned short* __restrict__ O) {
  __shared__ float s1[256], s2[256];
  const int row = blockIdx.x;
  const int tid = threadIdx.x;
  const float* x = X + (size_t)row * kDIM;
  float a0 = 0.f, a1 = 0.f;
  for (int i = tid; i < kDIM; i += 256) { float v = x[i]; a0 += v; a1 += v * v; }
  s1[tid] = a0; s2[tid] = a1;
  __syncthreads();
  for (int st = 128; st > 0; st >>= 1) {
    if (tid < st) { s1[tid] += s1[tid + st]; s2[tid] += s2[tid + st]; }
    __syncthreads();
  }
  const float mu  = s1[0] * (1.f / kDIM);
  const float var = s2[0] * (1.f / kDIM) - mu * mu;
  const float rs  = rsqrtf(var + 1e-5f);
  for (int i = tid; i < kDIM; i += 256)
    O[(size_t)row * kDIM + i] = f2bf((x[i] - mu) * rs * g[i] + b[i]);
}

// ---------------- WMMA GEMM kernels ----------------
// C(MxN) = A(MxK bf16 row-major) @ Bt(NxK bf16 row-major)^T
// Full-tile kernels: 8 waves/block, block tile 128x128, 64x32 per wave
// (acc[4][2]); A tile (128x32) staged in LDS via async copy, double-buffered,
// one barrier per K-step, last step peeled (no branches in the hot loop).

__global__ __launch_bounds__(256) void gemm_bf16_kernel(
    const unsigned short* __restrict__ A, const unsigned short* __restrict__ Bt,
    unsigned short* __restrict__ C, int M, int N, int K) {
  __shared__ unsigned short ldsA[2][128 * 32];
  const int tid   = threadIdx.x;
  const int wave  = tid >> 5;
  const int wm    = wave >> 2;               // 0..1
  const int wn    = wave & 3;                // 0..3
  const int rowblk = blockIdx.y * 128;
  const int row0   = rowblk + wm * 64;
  const int col0   = blockIdx.x * 128 + wn * 32;
  const int cr = tid >> 1;                   // copy row 0..127
  const int ch = (tid & 1) * 16;             // copy half (elements)
  v8f acc[4][2];
#pragma unroll
  for (int i = 0; i < 4; ++i)
#pragma unroll
    for (int j = 0; j < 2; ++j) acc[i][j] = v8f_zero();

  const unsigned loff0 = lds_off_of(&ldsA[0][cr * 32 + ch]);
  const unsigned loff1 = lds_off_of(&ldsA[1][cr * 32 + ch]);
  const unsigned short* gp = A + (size_t)(rowblk + cr) * K + ch;

  auto compute = [&](int cur, int kb) {
    v16bf a[4];
#pragma unroll
    for (int i = 0; i < 4; ++i)
      a[i] = load_frag_fast(&ldsA[cur][(wm * 64 + 16 * i) * 32], 32);
    v16bf b0 = load_frag_fast(Bt + (size_t)col0 * K + kb,        K);
    v16bf b1 = load_frag_fast(Bt + (size_t)(col0 + 16) * K + kb, K);
#pragma unroll
    for (int i = 0; i < 4; ++i) {
      acc[i][0] = wmma_bf16(a[i], b0, acc[i][0]);
      acc[i][1] = wmma_bf16(a[i], b1, acc[i][1]);
    }
  };

  async_copy_b128(loff0, gp);
  async_copy_b128(loff0 + 16, gp + 8);
  gp += 32;
  const int nsteps = K / 32;
  for (int s = 0; s < nsteps - 1; ++s) {
    wait_async0();
    __syncthreads();
    unsigned ln = ((s & 1) == 0) ? loff1 : loff0;
    async_copy_b128(ln, gp);
    async_copy_b128(ln + 16, gp + 8);
    gp += 32;
    compute(s & 1, s * 32);
  }
  wait_async0();
  __syncthreads();
  compute((nsteps - 1) & 1, (nsteps - 1) * 32);

  const int lane = threadIdx.x & 31, hf = lane >> 4, nl = lane & 15;
#pragma unroll
  for (int i = 0; i < 4; ++i)
#pragma unroll
    for (int j = 0; j < 2; ++j)
#pragma unroll
      for (int r = 0; r < 8; ++r) {
        int m = row0 + 16 * i + r + 8 * hf;
        int c = col0 + 16 * j + nl;
        C[(size_t)m * N + c] = f2bf(acc[i][j][r]);
      }
}

// f32 out, + bias[col] + residual (full tiles only)
__global__ __launch_bounds__(256) void gemm_f32_kernel(
    const unsigned short* __restrict__ A, const unsigned short* __restrict__ Bt,
    const float* __restrict__ bias, const float* __restrict__ resid,
    float* __restrict__ C, int M, int N, int K) {
  __shared__ unsigned short ldsA[2][128 * 32];
  const int tid   = threadIdx.x;
  const int wave  = tid >> 5;
  const int wm    = wave >> 2;
  const int wn    = wave & 3;
  const int rowblk = blockIdx.y * 128;
  const int row0   = rowblk + wm * 64;
  const int col0   = blockIdx.x * 128 + wn * 32;
  const int cr = tid >> 1;
  const int ch = (tid & 1) * 16;
  v8f acc[4][2];
#pragma unroll
  for (int i = 0; i < 4; ++i)
#pragma unroll
    for (int j = 0; j < 2; ++j) acc[i][j] = v8f_zero();

  const unsigned loff0 = lds_off_of(&ldsA[0][cr * 32 + ch]);
  const unsigned loff1 = lds_off_of(&ldsA[1][cr * 32 + ch]);
  const unsigned short* gp = A + (size_t)(rowblk + cr) * K + ch;

  auto compute = [&](int cur, int kb) {
    v16bf a[4];
#pragma unroll
    for (int i = 0; i < 4; ++i)
      a[i] = load_frag_fast(&ldsA[cur][(wm * 64 + 16 * i) * 32], 32);
    v16bf b0 = load_frag_fast(Bt + (size_t)col0 * K + kb,        K);
    v16bf b1 = load_frag_fast(Bt + (size_t)(col0 + 16) * K + kb, K);
#pragma unroll
    for (int i = 0; i < 4; ++i) {
      acc[i][0] = wmma_bf16(a[i], b0, acc[i][0]);
      acc[i][1] = wmma_bf16(a[i], b1, acc[i][1]);
    }
  };

  async_copy_b128(loff0, gp);
  async_copy_b128(loff0 + 16, gp + 8);
  gp += 32;
  const int nsteps = K / 32;
  for (int s = 0; s < nsteps - 1; ++s) {
    wait_async0();
    __syncthreads();
    unsigned ln = ((s & 1) == 0) ? loff1 : loff0;
    async_copy_b128(ln, gp);
    async_copy_b128(ln + 16, gp + 8);
    gp += 32;
    compute(s & 1, s * 32);
  }
  wait_async0();
  __syncthreads();
  compute((nsteps - 1) & 1, (nsteps - 1) * 32);

  const int lane = threadIdx.x & 31, hf = lane >> 4, nl = lane & 15;
#pragma unroll
  for (int i = 0; i < 4; ++i)
#pragma unroll
    for (int j = 0; j < 2; ++j)
#pragma unroll
      for (int r = 0; r < 8; ++r) {
        int m = row0 + 16 * i + r + 8 * hf;
        int c = col0 + 16 * j + nl;
        C[(size_t)m * N + c] = acc[i][j][r] + bias[c] + resid[(size_t)m * N + c];
      }
}

// fused GEGLU: H = (A@ff1[:, :FFH] + b_x) * gelu(A@ff1[:, FFH:] + b_g), bf16 out
__global__ __launch_bounds__(256) void gemm_geglu_kernel(
    const unsigned short* __restrict__ A, const unsigned short* __restrict__ Bt, // (2*FFH) x DIM
    const float* __restrict__ bias, unsigned short* __restrict__ H, int M) {
  const int K = kDIM, N = kFFH;
  __shared__ unsigned short ldsA[2][128 * 32];
  const int tid   = threadIdx.x;
  const int wave  = tid >> 5;
  const int wm    = wave >> 2;
  const int wn    = wave & 3;
  const int rowblk = blockIdx.y * 128;
  const int row0   = rowblk + wm * 64;
  const int col0   = blockIdx.x * 128 + wn * 32;
  const int cr = tid >> 1;
  const int ch = (tid & 1) * 16;
  v8f accx[4][2], accg[4][2];
#pragma unroll
  for (int i = 0; i < 4; ++i)
#pragma unroll
    for (int j = 0; j < 2; ++j) { accx[i][j] = v8f_zero(); accg[i][j] = v8f_zero(); }

  const unsigned loff0 = lds_off_of(&ldsA[0][cr * 32 + ch]);
  const unsigned loff1 = lds_off_of(&ldsA[1][cr * 32 + ch]);
  const unsigned short* gp = A + (size_t)(rowblk + cr) * K + ch;

  auto compute = [&](int cur, int kb) {
    v16bf a[4];
#pragma unroll
    for (int i = 0; i < 4; ++i)
      a[i] = load_frag_fast(&ldsA[cur][(wm * 64 + 16 * i) * 32], 32);
    v16bf bx0 = load_frag_fast(Bt + (size_t)col0 * K + kb,            K);
    v16bf bx1 = load_frag_fast(Bt + (size_t)(col0 + 16) * K + kb,     K);
    v16bf bg0 = load_frag_fast(Bt + (size_t)(col0 + N) * K + kb,      K);
    v16bf bg1 = load_frag_fast(Bt + (size_t)(col0 + N + 16) * K + kb, K);
#pragma unroll
    for (int i = 0; i < 4; ++i) {
      accx[i][0] = wmma_bf16(a[i], bx0, accx[i][0]);
      accx[i][1] = wmma_bf16(a[i], bx1, accx[i][1]);
      accg[i][0] = wmma_bf16(a[i], bg0, accg[i][0]);
      accg[i][1] = wmma_bf16(a[i], bg1, accg[i][1]);
    }
  };

  async_copy_b128(loff0, gp);
  async_copy_b128(loff0 + 16, gp + 8);
  gp += 32;
  const int nsteps = K / 32;
  for (int s = 0; s < nsteps - 1; ++s) {
    wait_async0();
    __syncthreads();
    unsigned ln = ((s & 1) == 0) ? loff1 : loff0;
    async_copy_b128(ln, gp);
    async_copy_b128(ln + 16, gp + 8);
    gp += 32;
    compute(s & 1, s * 32);
  }
  wait_async0();
  __syncthreads();
  compute((nsteps - 1) & 1, (nsteps - 1) * 32);

  const int lane = threadIdx.x & 31, hf = lane >> 4, nl = lane & 15;
#pragma unroll
  for (int i = 0; i < 4; ++i)
#pragma unroll
    for (int j = 0; j < 2; ++j)
#pragma unroll
      for (int r = 0; r < 8; ++r) {
        int m = row0 + 16 * i + r + 8 * hf;
        int c = col0 + 16 * j + nl;
        float xv = accx[i][j][r] + bias[c];
        float gv = accg[i][j][r] + bias[c + N];
        float ge = 0.5f * gv * (1.f + erff(gv * 0.70710678118654752f));
        H[(size_t)m * N + c] = f2bf(xv * ge);
      }
}

// Guarded GEMM for the encoder projections (M=616): direct loads, old tiling.
__device__ inline v16bf gemm_load_a_g(const unsigned short* A, int base, int M, int K, int kb) {
  int bc = (base < M) ? base : ((M > 16) ? M - 16 : 0);   // clamped, always valid
  if (M - base >= 16) return load_frag_fast(A + (size_t)bc * K + kb, K);
  return load_frag_guard(A + (size_t)bc * K + kb, K, M - base, 32);
}

__global__ __launch_bounds__(256) void gemm_bf16_guard_kernel(
    const unsigned short* __restrict__ A, const unsigned short* __restrict__ Bt,
    unsigned short* __restrict__ C, int M, int N, int K) {
  const int wave = threadIdx.x >> 5;
  const int wm = wave >> 1, wn = wave & 1;
  const int row0 = blockIdx.y * 128 + wm * 32;
  const int col0 = blockIdx.x * 64 + wn * 32;
  v8f acc[2][2];
#pragma unroll
  for (int i = 0; i < 2; ++i)
#pragma unroll
    for (int j = 0; j < 2; ++j) acc[i][j] = v8f_zero();
  for (int kb = 0; kb < K; kb += 32) {
    v16bf a0 = gemm_load_a_g(A, row0,      M, K, kb);
    v16bf a1 = gemm_load_a_g(A, row0 + 16, M, K, kb);
    v16bf b0 = load_frag_fast(Bt + (size_t)col0 * K + kb,        K);
    v16bf b1 = load_frag_fast(Bt + (size_t)(col0 + 16) * K + kb, K);
    acc[0][0] = wmma_bf16(a0, b0, acc[0][0]);
    acc[0][1] = wmma_bf16(a0, b1, acc[0][1]);
    acc[1][0] = wmma_bf16(a1, b0, acc[1][0]);
    acc[1][1] = wmma_bf16(a1, b1, acc[1][1]);
  }
  const int lane = threadIdx.x & 31, hf = lane >> 4, nl = lane & 15;
#pragma unroll
  for (int i = 0; i < 2; ++i)
#pragma unroll
    for (int j = 0; j < 2; ++j)
#pragma unroll
      for (int r = 0; r < 8; ++r) {
        int m = row0 + 16 * i + r + 8 * hf;
        int c = col0 + 16 * j + nl;
        if (m < M && c < N) C[(size_t)m * N + c] = f2bf(acc[i][j][r]);
      }
}

// ---------------- flash attention (wave32, 16 queries per wave) ----------------
template <bool GUARD>
__device__ void flash_seg(const unsigned short* Kseg, const unsigned short* Vt, int ldv,
                          int nk, int nkpad, const v16bf* qf, float scale,
                          v8f* o, float* m_, float* l_, unsigned short* ldsP) {
  const int lane = threadIdx.x & 31;
  const int hf   = lane >> 4;
  const int nl   = lane & 15;
  for (int kb = 0; kb < nkpad; kb += 32) {
    const int kb1  = kb + 16;
    const int kb1c = (!GUARD || kb1 + 16 <= nk) ? kb1 : (nk - 16);
    v8f s0 = v8f_zero(), s1 = v8f_zero();
#pragma unroll
    for (int c = 0; c < 3; ++c) {
      const unsigned short* p0 = Kseg + (size_t)kb * kDIM + 32 * c;
      const unsigned short* p1 = Kseg + (size_t)kb1c * kDIM + 32 * c;
      v16bf bk0, bk1;
      if (c < 2) {
        bk0 = GUARD ? load_frag_guard(p0, kDIM, nk - kb, 32)  : load_frag_fast(p0, kDIM);
        bk1 = GUARD ? load_frag_guard(p1, kDIM, nk - kb1, 32) : load_frag_fast(p1, kDIM);
      } else {
        bk0 = GUARD ? load_frag_guard(p0, kDIM, nk - kb, 16)  : load_frag_c2(p0, kDIM);
        bk1 = GUARD ? load_frag_guard(p1, kDIM, nk - kb1, 16) : load_frag_c2(p1, kDIM);
      }
      s0 = wmma_bf16(qf[c], bk0, s0);
      s1 = wmma_bf16(qf[c], bk1, s1);
    }
    const bool msk0 = GUARD && ((kb + nl)      >= nk);
    const bool msk1 = GUARD && ((kb + 16 + nl) >= nk);
#pragma unroll
    for (int r = 0; r < 8; ++r) {
      float v0 = msk0 ? -1e30f : s0[r] * scale;
      float v1 = msk1 ? -1e30f : s1[r] * scale;
      float tmax = rmax16(fmaxf(v0, v1));
      float newm = fmaxf(m_[r], tmax);
      float alpha = __expf(m_[r] - newm);
      float p0 = msk0 ? 0.f : __expf(v0 - newm);
      float p1 = msk1 ? 0.f : __expf(v1 - newm);
      l_[r] = l_[r] * alpha + rsum16(p0) + rsum16(p1);
      m_[r] = newm;
#pragma unroll
      for (int t = 0; t < 5; ++t) o[t][r] *= alpha;
      int mr = r + 8 * hf;
      ldsP[mr * 32 + nl]      = f2bf(p0);
      ldsP[mr * 32 + 16 + nl] = f2bf(p1);
    }
    asm volatile("s_wait_dscnt 0" ::: "memory");   // LDS store -> load ordering
    v16bf ap = load_frag_fast(ldsP, 32);
#pragma unroll
    for (int t = 0; t < 5; ++t) {
      v16bf bv = load_frag_fast(Vt + (size_t)(t * 16) * ldv + kb, ldv);
      o[t] = wmma_bf16(ap, bv, o[t]);
    }
  }
}

__device__ inline void attn_store(unsigned short* Op, const v8f* o, const float* l_) {
  const int lane = threadIdx.x & 31, hf = lane >> 4, nl = lane & 15;
#pragma unroll
  for (int t = 0; t < 5; ++t)
#pragma unroll
    for (int r = 0; r < 8; ++r) {
      int mr = r + 8 * hf;
      Op[(size_t)mr * kDIM + t * 16 + nl] = f2bf(o[t][r] / l_[r]);
    }
}

// self attention with sparse-causal KV: keys = [frame 0 ; frame max(f-1,0)]
__global__ __launch_bounds__(256) void attn_self_kernel(
    const unsigned short* __restrict__ Q, const unsigned short* __restrict__ K,
    const unsigned short* __restrict__ Vt, unsigned short* __restrict__ Out,
    const int* __restrict__ vlen_p) {
  __shared__ unsigned short ldsP[8 * 16 * 32];
  const int wave = threadIdx.x >> 5;
  const int bf   = blockIdx.z;
  const int head = blockIdx.y;
  const int qt   = blockIdx.x * 8 + wave;
  const int vlen = vlen_p[0];
  const int bidx = bf / vlen;
  const int fr   = bf - bidx * vlen;
  const int f0   = bidx * vlen;
  const int f1   = bidx * vlen + (fr > 0 ? fr - 1 : 0);
  const int qrow = bf * kSEQ + qt * 16;
  const unsigned short* Qp = Q + (size_t)qrow * kDIM + head * kHD;
  v16bf qf[3];
  qf[0] = load_frag_fast(Qp,      kDIM);
  qf[1] = load_frag_fast(Qp + 32, kDIM);
  qf[2] = load_frag_c2(Qp + 64,   kDIM);
  v8f o[5]; float m_[8], l_[8];
#pragma unroll
  for (int t = 0; t < 5; ++t) o[t] = v8f_zero();
#pragma unroll
  for (int r = 0; r < 8; ++r) { m_[r] = -1e30f; l_[r] = 0.f; }
  const float scale = rsqrtf((float)kHD);
  unsigned short* lp = ldsP + wave * (16 * 32);
  flash_seg<false>(K  + (size_t)f0 * kSEQ * kDIM + head * kHD,
                   Vt + ((size_t)f0 * kHEADS + head) * kHD * kSEQ, kSEQ,
                   kSEQ, kSEQ, qf, scale, o, m_, l_, lp);
  flash_seg<false>(K  + (size_t)f1 * kSEQ * kDIM + head * kHD,
                   Vt + ((size_t)f1 * kHEADS + head) * kHD * kSEQ, kSEQ,
                   kSEQ, kSEQ, qf, scale, o, m_, l_, lp);
  attn_store(Out + (size_t)qrow * kDIM + head * kHD, o, l_);
}

// cross attention over kCTX keys (padded to 96, masked)
__global__ __launch_bounds__(256) void attn_cross_kernel(
    const unsigned short* __restrict__ Q, const unsigned short* __restrict__ K,
    const unsigned short* __restrict__ Vt, unsigned short* __restrict__ Out) {
  __shared__ unsigned short ldsP[8 * 16 * 32];
  const int wave = threadIdx.x >> 5;
  const int bf   = blockIdx.z;
  const int head = blockIdx.y;
  const int qt   = blockIdx.x * 8 + wave;
  const int qrow = bf * kSEQ + qt * 16;
  const unsigned short* Qp = Q + (size_t)qrow * kDIM + head * kHD;
  v16bf qf[3];
  qf[0] = load_frag_fast(Qp,      kDIM);
  qf[1] = load_frag_fast(Qp + 32, kDIM);
  qf[2] = load_frag_c2(Qp + 64,   kDIM);
  v8f o[5]; float m_[8], l_[8];
#pragma unroll
  for (int t = 0; t < 5; ++t) o[t] = v8f_zero();
#pragma unroll
  for (int r = 0; r < 8; ++r) { m_[r] = -1e30f; l_[r] = 0.f; }
  const float scale = rsqrtf((float)kHD);
  unsigned short* lp = ldsP + wave * (16 * 32);
  flash_seg<true>(K  + (size_t)bf * kCTX * kDIM + head * kHD,
                  Vt + ((size_t)bf * kHEADS + head) * kHD * kLDVC, kLDVC,
                  kCTX, 96, qf, scale, o, m_, l_, lp);
  attn_store(Out + (size_t)qrow * kDIM + head * kHD, o, l_);
}

// ---------------- host-side launch ----------------
extern "C" void kernel_launch(void* const* d_in, const int* in_sizes, int n_in,
                              void* d_out, int out_size, void* d_ws, size_t ws_size,
                              hipStream_t stream) {
  (void)in_sizes; (void)n_in; (void)out_size; (void)ws_size;
  const float* hs0   = (const float*)d_in[0];
  const float* enc   = (const float*)d_in[1];
  const float* n1_g  = (const float*)d_in[2];
  const float* n1_b  = (const float*)d_in[3];
  const float* q1    = (const float*)d_in[4];
  const float* k1    = (const float*)d_in[5];
  const float* v1    = (const float*)d_in[6];
  const float* o1    = (const float*)d_in[7];
  const float* o1_b  = (const float*)d_in[8];
  const float* n2_g  = (const float*)d_in[9];
  const float* n2_b  = (const float*)d_in[10];
  const float* q2    = (const float*)d_in[11];
  const float* k2    = (const float*)d_in[12];
  const float* v2    = (const float*)d_in[13];
  const float* o2    = (const float*)d_in[14];
  const float* o2_b  = (const float*)d_in[15];
  const float* n3_g  = (const float*)d_in[16];
  const float* n3_b  = (const float*)d_in[17];
  const float* ff1   = (const float*)d_in[18];
  const float* ff1_b = (const float*)d_in[19];
  const float* ff2   = (const float*)d_in[20];
  const float* ff2_b = (const float*)d_in[21];
  const int*   vlen  = (const int*)d_in[22];
  float* out = (float*)d_out;

  char* ws = (char*)d_ws;
  size_t off = 0;
  auto alloc = [&](size_t bytes) -> void* {
    void* p = ws + off;
    off += (bytes + 255) & ~(size_t)255;
    return p;
  };
  float* hs1 = (float*)alloc((size_t)kROWS * kDIM * 4);
  float* hs2 = (float*)alloc((size_t)kROWS * kDIM * 4);
  unsigned short* nbf   = (unsigned short*)alloc((size_t)kROWS * kDIM * 2);
  unsigned short* qbf   = (unsigned short*)alloc((size_t)kROWS * kDIM * 2);
  unsigned short* kbf   = (unsigned short*)alloc((size_t)kROWS * kDIM * 2);
  unsigned short* vbf   = (unsigned short*)alloc((size_t)kROWS * kDIM * 2);
  unsigned short* abf   = (unsigned short*)alloc((size_t)kROWS * kDIM * 2);
  unsigned short* vtbf  = (unsigned short*)alloc((size_t)kROWS * kDIM * 2);   // transposed self V
  unsigned short* vct   = (unsigned short*)alloc((size_t)kBF * kHEADS * kHD * kLDVC * 2);
  unsigned short* encbf = (unsigned short*)alloc((size_t)kBF * kCTX * kCROSS * 2);
  unsigned short* hbf   = (unsigned short*)alloc((size_t)kROWS * kFFH * 2);
  unsigned short* wq1t  = (unsigned short*)alloc((size_t)kDIM * kDIM * 2);
  unsigned short* wk1t  = (unsigned short*)alloc((size_t)kDIM * kDIM * 2);
  unsigned short* wv1t  = (unsigned short*)alloc((size_t)kDIM * kDIM * 2);
  unsigned short* wo1t  = (unsigned short*)alloc((size_t)kDIM * kDIM * 2);
  unsigned short* wq2t  = (unsigned short*)alloc((size_t)kDIM * kDIM * 2);
  unsigned short* wo2t  = (unsigned short*)alloc((size_t)kDIM * kDIM * 2);
  unsigned short* wk2t  = (unsigned short*)alloc((size_t)kDIM * kCROSS * 2);
  unsigned short* wv2t  = (unsigned short*)alloc((size_t)kDIM * kCROSS * 2);
  unsigned short* wff1t = (unsigned short*)alloc((size_t)2 * kFFH * kDIM * 2);
  unsigned short* wff2t = (unsigned short*)alloc((size_t)kDIM * kFFH * 2);

  auto cdiv = [](int a, int b) { return (a + b - 1) / b; };

  // weight conversions (W is KxN -> Wt NxK bf16)
  wt_kernel<<<cdiv(kDIM * kDIM, 256), 256, 0, stream>>>(q1, wq1t, kDIM, kDIM);
  wt_kernel<<<cdiv(kDIM * kDIM, 256), 256, 0, stream>>>(k1, wk1t, kDIM, kDIM);
  wt_kernel<<<cdiv(kDIM * kDIM, 256), 256, 0, stream>>>(v1, wv1t, kDIM, kDIM);
  wt_kernel<<<cdiv(kDIM * kDIM, 256), 256, 0, stream>>>(o1, wo1t, kDIM, kDIM);
  wt_kernel<<<cdiv(kDIM * kDIM, 256), 256, 0, stream>>>(q2, wq2t, kDIM, kDIM);
  wt_kernel<<<cdiv(kDIM * kDIM, 256), 256, 0, stream>>>(o2, wo2t, kDIM, kDIM);
  wt_kernel<<<cdiv(kCROSS * kDIM, 256), 256, 0, stream>>>(k2, wk2t, kCROSS, kDIM);
  wt_kernel<<<cdiv(kCROSS * kDIM, 256), 256, 0, stream>>>(v2, wv2t, kCROSS, kDIM);
  wt_kernel<<<cdiv(kDIM * 2 * kFFH, 256), 256, 0, stream>>>(ff1, wff1t, kDIM, 2 * kFFH);
  wt_kernel<<<cdiv(kFFH * kDIM, 256), 256, 0, stream>>>(ff2, wff2t, kFFH, kDIM);
  cvt_kernel<<<cdiv(kBF * kCTX * kCROSS, 256), 256, 0, stream>>>(enc, encbf, kBF * kCTX * kCROSS);

  dim3 g640(kDIM / 128, kROWS / 128);                 // full-tile GEMMs, N=640
  dim3 gkv(cdiv(kDIM, 64), cdiv(kBF * kCTX, 128));    // guarded encoder GEMM
  dim3 gff1(kFFH / 128, kROWS / 128);                 // GEGLU
  dim3 ga(kSEQ / 16 / 8, kHEADS, kBF);

  // --- stage 1: self attention ---
  ln_kernel<<<kROWS, 256, 0, stream>>>(hs0, n1_g, n1_b, nbf);
  gemm_bf16_kernel<<<g640, 256, 0, stream>>>(nbf, wq1t, qbf, kROWS, kDIM, kDIM);
  gemm_bf16_kernel<<<g640, 256, 0, stream>>>(nbf, wk1t, kbf, kROWS, kDIM, kDIM);
  gemm_bf16_kernel<<<g640, 256, 0, stream>>>(nbf, wv1t, vbf, kROWS, kDIM, kDIM);
  vt_self_kernel<<<cdiv(kROWS * kDIM, 256), 256, 0, stream>>>(vbf, vtbf);
  attn_self_kernel<<<ga, 256, 0, stream>>>(qbf, kbf, vtbf, abf, vlen);
  gemm_f32_kernel<<<g640, 256, 0, stream>>>(abf, wo1t, o1_b, hs0, hs1, kROWS, kDIM, kDIM);

  // --- stage 2: cross attention ---
  ln_kernel<<<kROWS, 256, 0, stream>>>(hs1, n2_g, n2_b, nbf);
  gemm_bf16_kernel<<<g640, 256, 0, stream>>>(nbf, wq2t, qbf, kROWS, kDIM, kDIM);
  gemm_bf16_guard_kernel<<<gkv, 256, 0, stream>>>(encbf, wk2t, kbf, kBF * kCTX, kDIM, kCROSS);
  gemm_bf16_guard_kernel<<<gkv, 256, 0, stream>>>(encbf, wv2t, vbf, kBF * kCTX, kDIM, kCROSS);
  vt_cross_kernel<<<cdiv(kBF * kHEADS * kHD * kLDVC, 256), 256, 0, stream>>>(vbf, vct);
  attn_cross_kernel<<<ga, 256, 0, stream>>>(qbf, kbf, vct, abf);
  gemm_f32_kernel<<<g640, 256, 0, stream>>>(abf, wo2t, o2_b, hs1, hs2, kROWS, kDIM, kDIM);

  // --- stage 3: GEGLU FFN ---
  ln_kernel<<<kROWS, 256, 0, stream>>>(hs2, n3_g, n3_b, nbf);
  gemm_geglu_kernel<<<gff1, 256, 0, stream>>>(nbf, wff1t, ff1_b, hbf, kROWS);
  gemm_f32_kernel<<<g640, 256, 0, stream>>>(hbf, wff2t, ff2_b, hs2, out, kROWS, kDIM, kFFH);
}
